// DKD_47648367182498
// MI455X (gfx1250) — compile-verified, matrix operand order checked by reading, not run on hardware
//
#include <hip/hip_runtime.h>
#include <hip/hip_bf16.h>

typedef __attribute__((ext_vector_type(16))) _Float16 v16h;
typedef __attribute__((ext_vector_type(8)))  float    v8f;

#define H 2048
#define W 2048
#define NTILES (512*512)          // 262144 tile maxima
#define TOPK 2048
#define CAND_CAP 32768            // candidate cap for 1-block LDS sort (256KB LDS)
#define NHIST 65536               // 16-bit radix key (float bits >> 16)

// d_out layout (floats): kp [0,4096) | desc [4096,135168) | vals [135168,137216)
#define OUT_DESC_OFF 4096
#define OUT_VAL_OFF  135168

// ---------------- K0: zero histogram + meta ----------------
__global__ void dkd_zero(unsigned* __restrict__ hist, unsigned* __restrict__ meta) {
    int id = blockIdx.x * 256 + threadIdx.x;
    if (id < NHIST) hist[id] = 0u;
    if (id < 8)     meta[id] = 0u;
}

// ---------------- K1: per-4x4-tile max (+border zero) + radix histogram ----------------
__global__ void dkd_tilemax(const float* __restrict__ scores,
                            float* __restrict__ vals,
                            unsigned* __restrict__ coords,
                            unsigned* __restrict__ hist) {
    int id = blockIdx.x * blockDim.x + threadIdx.x;      // 0..262143
    int tr = id >> 9, tc = id & 511;
    int r0 = tr << 2, c0 = tc << 2;
    const float* p = scores + (size_t)r0 * W + c0;
    __builtin_prefetch(p + 4 * W, 0, 0);                 // global_prefetch_b8
    float best = -1.0f;
    int barg = 0;
#pragma unroll
    for (int ki = 0; ki < 4; ++ki) {
        float4 v4 = *(const float4*)(p + (size_t)ki * W);
        int r = r0 + ki;
        bool rowOK = (r >= 3) && (r <= H - 4);
        float e[4] = {v4.x, v4.y, v4.z, v4.w};
#pragma unroll
        for (int kj = 0; kj < 4; ++kj) {
            int c = c0 + kj;
            float val = (rowOK && c >= 3 && c <= W - 4) ? e[kj] : 0.0f;
            if (val > best) { best = val; barg = ki * 4 + kj; }
        }
    }
    int row = r0 + (barg >> 2);
    int col = c0 + (barg & 3);
    vals[id]   = best;
    coords[id] = ((unsigned)row << 12) | (unsigned)col;
    atomicAdd(&hist[__float_as_uint(best) >> 16], 1u);   // monotone key: vals >= 0
}

// ---------------- K2: find threshold key T = max key with count(key>=T) >= TOPK ----------
__global__ void dkd_thresh(const unsigned* __restrict__ hist, unsigned* __restrict__ meta) {
    __shared__ unsigned csum[1024];
    int t = threadIdx.x;
    unsigned s = 0;
#pragma unroll 4
    for (int i = 0; i < 64; ++i) s += hist[t * 64 + i];
    csum[t] = s;
    __syncthreads();
    if (t == 0) {
        unsigned cum = 0; int T = 0;
        for (int c = 1023; c >= 0; --c) {
            if (cum + csum[c] >= (unsigned)TOPK) {
                for (int k = c * 64 + 63; k >= c * 64; --k) {
                    cum += hist[k];
                    if (cum >= (unsigned)TOPK) { T = k; break; }
                }
                break;
            }
            cum += csum[c];
        }
        meta[0] = (unsigned)T;
    }
}

// ---------------- K3: compact candidates with key >= T ----------------
__global__ void dkd_compact(const float* __restrict__ vals,
                            const unsigned* __restrict__ coords,
                            const unsigned* __restrict__ meta,
                            float* __restrict__ cval,
                            unsigned* __restrict__ ccoord,
                            unsigned* __restrict__ counter) {
    int id = blockIdx.x * blockDim.x + threadIdx.x;
    float v = vals[id];
    if ((__float_as_uint(v) >> 16) >= meta[0]) {
        unsigned pos = atomicAdd(counter, 1u);
        if (pos < (unsigned)CAND_CAP) { cval[pos] = v; ccoord[pos] = coords[id]; }
    }
}

// ---------------- K4: single-WGP 32K bitonic sort in 256KB LDS, emit top-2048 ----------
__global__ void dkd_sort_emit(const float* __restrict__ cval,
                              const unsigned* __restrict__ ccoord,
                              const unsigned* __restrict__ counter,
                              float* __restrict__ out,
                              unsigned* __restrict__ topcoord) {
    extern __shared__ unsigned char sm[];
    float*    sv = (float*)sm;
    unsigned* sc = (unsigned*)(sm + (size_t)CAND_CAP * sizeof(float));
    int t = threadIdx.x;
    unsigned n = *counter; if (n > (unsigned)CAND_CAP) n = (unsigned)CAND_CAP;
    for (int i = t; i < CAND_CAP; i += 1024) {
        if (i < (int)n) { sv[i] = cval[i]; sc[i] = ccoord[i]; }
        else            { sv[i] = -1.0f;   sc[i] = 0u; }
    }
    __syncthreads();
    // bitonic sort, descending by value
    for (int k = 2; k <= CAND_CAP; k <<= 1) {
        for (int j = k >> 1; j > 0; j >>= 1) {
            for (int i = t; i < CAND_CAP; i += 1024) {
                int ixj = i ^ j;
                if (ixj > i) {
                    float a = sv[i], b = sv[ixj];
                    bool descBlock = ((i & k) == 0);
                    if (descBlock ? (a < b) : (a > b)) {
                        unsigned ca = sc[i], cb = sc[ixj];
                        sv[i] = b; sv[ixj] = a; sc[i] = cb; sc[ixj] = ca;
                    }
                }
            }
            __syncthreads();
        }
    }
    for (int kpt = t; kpt < TOPK; kpt += 1024) {
        float v = sv[kpt];
        unsigned cd = sc[kpt];
        unsigned row = cd >> 12, col = cd & 0xFFFu;
        out[kpt * 2 + 0]       = (float)col;   // x
        out[kpt * 2 + 1]       = (float)row;   // y
        out[OUT_VAL_OFF + kpt] = v;
        topcoord[kpt] = cd;
    }
}

// ---------------- K5: descriptor gather + WMMA sum-of-squares + normalize --------------
// One wave = 16 keypoints. A (16x32, f16) = squared descriptors laid out per the
// CDNA5 16-bit A-matrix pattern; B = all-ones so D[m,n] = rowsum(m) (layout-free).
// Two chained v_wmma_f32_16x16x32_f16 cover channels 0..63 with f32 accumulation.
__global__ void dkd_desc(const float* __restrict__ dmap,
                         const unsigned* __restrict__ topcoord,
                         float* __restrict__ outdesc) {
    int lane = threadIdx.x & 31;
    int wave = (blockIdx.x * blockDim.x + threadIdx.x) >> 5;   // 0..127
    int m    = lane & 15;
    int kp   = wave * 16 + m;
    unsigned cd = topcoord[kp];
    unsigned row = cd >> 12, col = cd & 0xFFFu;
    const float* base = dmap + (size_t)row * W + col;
    const size_t cstride = (size_t)H * W;

    // channel map for this lane's 8 A-VGPRs (16-bit A 16x32 layout)
    int kbase = (lane < 16) ? 0 : 8;
    int ch[16];
#pragma unroll
    for (int v = 0; v < 8; ++v) {
        int k0 = ((v < 4) ? (2 * v) : (16 + 2 * (v - 4))) + kbase;
        ch[2 * v] = k0; ch[2 * v + 1] = k0 + 1;
    }

    float d0[16], d1[16];
#pragma unroll
    for (int j = 0; j < 16; ++j) {
        d0[j] = base[cstride * (size_t)ch[j]];          // channels 0..31 (half A0)
        d1[j] = base[cstride * (size_t)(ch[j] + 32)];   // channels 32..63 (half A1)
    }

    v16h a0, a1, ones;
#pragma unroll
    for (int j = 0; j < 16; ++j) {
        a0[j]   = (_Float16)(d0[j] * d0[j]);
        a1[j]   = (_Float16)(d1[j] * d1[j]);
        ones[j] = (_Float16)1.0f;
    }
    v8f acc = {};
    acc = __builtin_amdgcn_wmma_f32_16x16x32_f16(false, a0, false, ones, (short)0, acc, false, false);
    acc = __builtin_amdgcn_wmma_f32_16x16x32_f16(false, a1, false, ones, (short)0, acc, false, false);

    // D layout: lanes 0-15 VGPR v -> rowsum(v); lanes 16-31 VGPR v -> rowsum(v+8).
    // Select component (lane&7), then ds_bpermute from the lane holding rowsum(lane&15).
    int s = lane & 7;
    float dsel = acc[0];
#pragma unroll
    for (int i = 1; i < 8; ++i) dsel = (s == i) ? acc[i] : dsel;
    int src = (lane & 7) | ((lane & 8) ? 16 : 0);
    float rsum = __int_as_float(__builtin_amdgcn_ds_bpermute(src << 2, __float_as_int(dsel)));
    float inv = rsqrtf(rsum);

    float* o = outdesc + (size_t)kp * 64;
#pragma unroll
    for (int j = 0; j < 16; ++j) {
        o[ch[j]]      = d0[j] * inv;
        o[ch[j] + 32] = d1[j] * inv;
    }
}

// ---------------- launch ----------------
extern "C" void kernel_launch(void* const* d_in, const int* in_sizes, int n_in,
                              void* d_out, int out_size, void* d_ws, size_t ws_size,
                              hipStream_t stream) {
    const float* scores = (const float*)d_in[0];   // (1,1,2048,2048) f32
    const float* dmap   = (const float*)d_in[1];   // (1,64,2048,2048) f32
    float* out = (float*)d_out;

    unsigned char* ws = (unsigned char*)d_ws;
    float*    vals     = (float*)(ws + 0);                       // 1 MB
    unsigned* coords   = (unsigned*)(ws + 1048576);              // 1 MB
    unsigned* hist     = (unsigned*)(ws + 2097152);              // 256 KB
    unsigned* meta     = (unsigned*)(ws + 2359296);              // 64 B  [0]=T,[1]=counter
    float*    cval     = (float*)(ws + 2359360);                 // 128 KB
    unsigned* ccoord   = (unsigned*)(ws + 2490432);              // 128 KB
    unsigned* topcoord = (unsigned*)(ws + 2621504);              // 8 KB

    dkd_zero   <<<256, 256, 0, stream>>>(hist, meta);
    dkd_tilemax<<<NTILES / 256, 256, 0, stream>>>(scores, vals, coords, hist);
    dkd_thresh <<<1, 1024, 0, stream>>>(hist, meta);
    dkd_compact<<<NTILES / 256, 256, 0, stream>>>(vals, coords, meta, cval, ccoord, meta + 1);
    dkd_sort_emit<<<1, 1024, (size_t)CAND_CAP * 8, stream>>>(cval, ccoord, meta + 1, out, topcoord);
    dkd_desc   <<<32, 128, 0, stream>>>(dmap, topcoord, out + OUT_DESC_OFF);
}
